// NTXentLoss_87789131530973
// MI455X (gfx1250) — matrix operand order repaired and checked by reading, not compiled, not run on hardware
//
#include <hip/hip_runtime.h>

// ---------------------------------------------------------------------------
// NT-Xent loss, fused for MI455X (gfx1250, wave32, WMMA + async global->LDS).
//   B = 4096 pairs, D = 256, 2B = 8192 total rows.
//   sim = Z Z^T (8192 x 8192 x 256) fused with exp-rowsum (logsumexp).
//   Precision: f16 hi/lo split (3 WMMA products) ~ f32-accurate cosine sims.
//   B tiles staged once per workgroup via GLOBAL_LOAD_ASYNC_TO_LDS_B128
//   (ASYNCcnt), double-buffered to overlap DMA with the WMMA stream.
// ---------------------------------------------------------------------------

typedef _Float16 h8   __attribute__((ext_vector_type(8)));
typedef _Float16 v16h __attribute__((ext_vector_type(16)));
typedef float    v8f  __attribute__((ext_vector_type(8)));

#define BROWS 4096
#define TWOB  8192
#define DDIM  256
#define NSLAB 8                         // column slabs (grid.y)
#define COLS_PER_SLAB (TWOB / NSLAB)    // 1024
#define NTILE (COLS_PER_SLAB / 16)      // 64 column tiles per block
#define LDS_STRIDE 264                  // halves; 528 B row pitch -> bank-free

__device__ __constant__ float kInvT = 14.2857142857142857f;  // 1/0.07

// ---------------------------------------------------------------------------
// Kernel 1: L2-normalize each of the 8192 rows; emit f16 hi + f16 lo residual.
// ---------------------------------------------------------------------------
__global__ __launch_bounds__(256) void ntx_normalize(
    const float* __restrict__ zi, const float* __restrict__ zj,
    _Float16* __restrict__ Zh, _Float16* __restrict__ Zl) {
  const int wave = threadIdx.x >> 5;
  const int lane = threadIdx.x & 31;
  const int row  = blockIdx.x * 8 + wave;          // 1024 blocks * 8 waves
  const float* src = (row < BROWS) ? (zi + (size_t)row * DDIM)
                                   : (zj + (size_t)(row - BROWS) * DDIM);
  float v[8];
  float ss = 0.0f;
#pragma unroll
  for (int j = 0; j < 8; ++j) {
    v[j] = src[lane * 8 + j];
    ss += v[j] * v[j];
  }
#pragma unroll
  for (int m = 16; m >= 1; m >>= 1) ss += __shfl_xor(ss, m, 32);
  const float inv = 1.0f / sqrtf(ss);
  _Float16* __restrict__ ph = Zh + (size_t)row * DDIM + lane * 8;
  _Float16* __restrict__ pl = Zl + (size_t)row * DDIM + lane * 8;
#pragma unroll
  for (int j = 0; j < 8; ++j) {
    const float n = v[j] * inv;
    const _Float16 h = (_Float16)n;
    ph[j] = h;
    pl[j] = (_Float16)(n - (float)h);
  }
}

// ---------------------------------------------------------------------------
// Kernel 2: pos[i] = dot(zi_n[i], zj_n[i]) in f32 (reconstructed hi+lo).
// ---------------------------------------------------------------------------
__global__ __launch_bounds__(256) void ntx_pos(
    const _Float16* __restrict__ Zh, const _Float16* __restrict__ Zl,
    float* __restrict__ pos) {
  const int wave = threadIdx.x >> 5;
  const int lane = threadIdx.x & 31;
  const int i = blockIdx.x * 8 + wave;             // 512 blocks * 8 waves
  const _Float16* ah = Zh + (size_t)i * DDIM + lane * 8;
  const _Float16* al = Zl + (size_t)i * DDIM + lane * 8;
  const _Float16* bh = Zh + (size_t)(i + BROWS) * DDIM + lane * 8;
  const _Float16* bl = Zl + (size_t)(i + BROWS) * DDIM + lane * 8;
  float acc = 0.0f;
#pragma unroll
  for (int j = 0; j < 8; ++j) {
    const float a = (float)ah[j] + (float)al[j];
    const float b = (float)bh[j] + (float)bl[j];
    acc = fmaf(a, b, acc);
  }
#pragma unroll
  for (int m = 16; m >= 1; m >>= 1) acc += __shfl_xor(acc, m, 32);
  if (lane == 0) pos[i] = acc;
}

// ---------------------------------------------------------------------------
// Combine two 16 B half8 chunks (at p and p+16) into a v16h fragment.
// ---------------------------------------------------------------------------
__device__ __forceinline__ v16h combine16(const _Float16* p) {
  const h8 a = *(const h8*)p;
  const h8 b = *(const h8*)(p + 16);
  return __builtin_shufflevector(a, b, 0, 1, 2, 3, 4, 5, 6, 7,
                                       8, 9, 10, 11, 12, 13, 14, 15);
}

// A-fragment (16-bit A 16x32): lane(m) = row M, khalf selects K-half;
// two 16 B chunks at K {kb..kb+7} and {kb+16..kb+23} (lanes 16-31: +8).
__device__ __forceinline__ v16h load_afrag(const _Float16* __restrict__ zrow,
                                           int kchunk, int khalf) {
  const h8 a = *(const h8*)(zrow + kchunk * 32 + khalf * 8);
  const h8 b = *(const h8*)(zrow + kchunk * 32 + 16 + khalf * 8);
  return __builtin_shufflevector(a, b, 0, 1, 2, 3, 4, 5, 6, 7,
                                       8, 9, 10, 11, 12, 13, 14, 15);
}

// ---------------------------------------------------------------------------
// Async global -> LDS, 16 bytes, tracked by ASYNCcnt (gfx1250).
// LDS offset = low 32 bits of the flat address (LDS aperture truncation).
// ---------------------------------------------------------------------------
__device__ __forceinline__ void async_ld16(unsigned lds_off, const void* gptr) {
  asm volatile("global_load_async_to_lds_b128 %0, %1, off"
               :: "v"(lds_off), "v"(gptr)
               : "memory");
}
__device__ __forceinline__ void wait_async0() {
  asm volatile("s_wait_asynccnt 0x0" ::: "memory");
}

// Stage one 16-column tile (hi+lo) into an LDS buffer: 1024 x 16 B chunks,
// 4 per thread, issued as async DMA (no VGPR landing, no s_wait needed here).
__device__ __forceinline__ void stage_tile(
    const _Float16* __restrict__ Zh, const _Float16* __restrict__ Zl,
    _Float16* bufH, _Float16* bufL, int colBase, int tid) {
  const int half = tid >> 7;            // 0 -> hi array, 1 -> lo array
  const int base = (tid & 127) * 4;     // 4 consecutive chunks, same column
  const _Float16* src = half ? Zl : Zh;
  _Float16* dst = half ? bufL : bufH;
  const int col   = base >> 5;          // 32 chunks of 16 B per 512 B row
  const int off8  = (base & 31) * 8;    // offset in halves
  const _Float16* g = src + (size_t)(colBase + col) * DDIM + off8;
  _Float16* l = dst + col * LDS_STRIDE + off8;
#pragma unroll
  for (int c = 0; c < 4; ++c)
    async_ld16((unsigned)(size_t)(l + c * 8), (const void*)(g + c * 8));
}

// ---------------------------------------------------------------------------
// Kernel 3: fused sim-GEMM + exp + row-sum.
//   grid = (64 row blocks of 128, 8 column slabs of 1024), 256 threads.
//   A frags (hi+lo, K=256) preloaded in 128 VGPRs per wave; B tiles staged
//   once per block in LDS via async DMA, double-buffered, conflict-free
//   ds_load_b128 consumption, 24 wmma per 16x16 tile, exp epilogue.
// ---------------------------------------------------------------------------
__global__ __launch_bounds__(256, 1) void ntx_simexp(
    const _Float16* __restrict__ Zh, const _Float16* __restrict__ Zl,
    float* __restrict__ rowPartial) {
  __shared__ __align__(64) _Float16 ldsB[2][2][16 * LDS_STRIDE];

  const int tid   = threadIdx.x;
  const int wave  = tid >> 5;
  const int lane  = tid & 31;
  const int m     = lane & 15;       // M (A/C rows) or N (B/C cols)
  const int khalf = lane >> 4;       // which K-half this lane carries
  const int rowBase = blockIdx.x * 128 + wave * 16;
  const int myrow   = rowBase + m;
  const int colSlab = blockIdx.y * COLS_PER_SLAB;

  // Preload this wave's A fragments for the whole K dimension (hi and lo).
  v16h ah[8], al[8];
  const _Float16* arow_h = Zh + (size_t)myrow * DDIM;
  const _Float16* arow_l = Zl + (size_t)myrow * DDIM;
#pragma unroll
  for (int k = 0; k < 8; ++k) {
    ah[k] = load_afrag(arow_h, k, khalf);
    al[k] = load_afrag(arow_l, k, khalf);
  }

  // Prime the pipeline: stage tile 0 into buffer 0.
  stage_tile(Zh, Zl, &ldsB[0][0][0], &ldsB[0][1][0], colSlab, tid);

  float racc[8] = {};

  for (int ct = 0; ct < NTILE; ++ct) {
    const int cur = ct & 1;
    // My async loads for buffer `cur` are the only outstanding ones.
    wait_async0();
    __syncthreads();   // everyone's DMA landed; prev readers of buf cur^1 done
    if (ct + 1 < NTILE)   // overlap next tile's DMA with this tile's WMMA
      stage_tile(Zh, Zl, &ldsB[cur ^ 1][0][0], &ldsB[cur ^ 1][1][0],
                 colSlab + (ct + 1) * 16, tid);

    // B fragment: lane(m)=col N; K {8*khalf..+7} and {16+8*khalf..+7}.
    const _Float16* brow_h = &ldsB[cur][0][m * LDS_STRIDE + khalf * 8];
    const _Float16* brow_l = &ldsB[cur][1][m * LDS_STRIDE + khalf * 8];

    v8f c = {};
#pragma unroll
    for (int k = 0; k < 8; ++k) {
      const v16h bh = combine16(brow_h + k * 32);
      const v16h bl = combine16(brow_l + k * 32);
      // sim += Al*Bh + Ah*Bl + Ah*Bh  (lo*lo dropped: below f32 ulp)
      c = __builtin_amdgcn_wmma_f32_16x16x32_f16(false, al[k], false, bh,
                                                 (short)0, c, false, false);
      c = __builtin_amdgcn_wmma_f32_16x16x32_f16(false, ah[k], false, bl,
                                                 (short)0, c, false, false);
      c = __builtin_amdgcn_wmma_f32_16x16x32_f16(false, ah[k], false, bh,
                                                 (short)0, c, false, false);
    }

    // Epilogue: shifted exp (max cosine = 1 -> arg <= 0), diagonal masked.
    const int colBase = colSlab + ct * 16;
    const int col = colBase + m;
#pragma unroll
    for (int i = 0; i < 8; ++i) {
      const int row = rowBase + i + (khalf << 3);
      float e = __expf((c[i] - 1.0f) * kInvT);
      e = (row == col) ? 0.0f : e;
      racc[i] += e;
    }
  }

  // Reduce each row's partial across the 16 lanes that share it.
#pragma unroll
  for (int i = 0; i < 8; ++i) {
    float s = racc[i];
#pragma unroll
    for (int msk = 1; msk < 16; msk <<= 1) s += __shfl_xor(s, msk, 16);
    racc[i] = s;
  }
  if (m == 0) {   // lane 0 -> rows +0..7 ; lane 16 -> rows +8..15
#pragma unroll
    for (int i = 0; i < 8; ++i) {
      const int row = rowBase + i + (khalf << 3);
      rowPartial[(size_t)blockIdx.y * TWOB + row] = racc[i];
    }
  }
}

// ---------------------------------------------------------------------------
// Kernel 4: per_row = log(sum_slabs) + 1/T - pos/T ; loss = mean(per_row).
// ---------------------------------------------------------------------------
__global__ __launch_bounds__(256) void ntx_finish(
    const float* __restrict__ rowPartial, const float* __restrict__ pos,
    float* __restrict__ out) {
  __shared__ float sacc[256];
  const int t = threadIdx.x;
  float acc = 0.0f;
  for (int r = t; r < TWOB; r += 256) {
    float s = 0.0f;
#pragma unroll
    for (int p = 0; p < NSLAB; ++p) s += rowPartial[p * TWOB + r];
    acc += logf(s) + kInvT - pos[r & (BROWS - 1)] * kInvT;
  }
  sacc[t] = acc;
  __syncthreads();
  for (int off = 128; off >= 1; off >>= 1) {
    if (t < off) sacc[t] += sacc[t + off];
    __syncthreads();
  }
  if (t == 0) out[0] = sacc[0] * (1.0f / (float)TWOB);
}

// ---------------------------------------------------------------------------
extern "C" void kernel_launch(void* const* d_in, const int* in_sizes, int n_in,
                              void* d_out, int out_size, void* d_ws, size_t ws_size,
                              hipStream_t stream) {
  (void)in_sizes; (void)n_in; (void)out_size; (void)ws_size;
  const float* zi = (const float*)d_in[0];   // (4096, 256) f32
  const float* zj = (const float*)d_in[1];   // (4096, 256) f32
  float* out = (float*)d_out;                // scalar f32

  // Workspace layout (~8.3 MiB):
  //   Zh : 8192*256 f16 = 4 MiB
  //   Zl : 8192*256 f16 = 4 MiB
  //   pos: 4096 f32     = 16 KiB
  //   rowPartial: 8*8192 f32 = 256 KiB   (fully overwritten -> no init)
  char* ws = (char*)d_ws;
  _Float16* Zh = (_Float16*)ws;
  _Float16* Zl = (_Float16*)(ws + (size_t)TWOB * DDIM * sizeof(_Float16));
  float* pos   = (float*)(ws + (size_t)TWOB * DDIM * sizeof(_Float16) * 2);
  float* rowPartial =
      (float*)(ws + (size_t)TWOB * DDIM * sizeof(_Float16) * 2 + BROWS * sizeof(float));

  ntx_normalize<<<TWOB / 8, 256, 0, stream>>>(zi, zj, Zh, Zl);
  ntx_pos<<<BROWS / 8, 256, 0, stream>>>(Zh, Zl, pos);
  dim3 grid(TWOB / 128, NSLAB);
  ntx_simexp<<<grid, 256, 0, stream>>>(Zh, Zl, rowPartial);
  ntx_finish<<<1, 256, 0, stream>>>(rowPartial, pos, out);
}
